// BarlowTwins_1056561955229
// MI455X (gfx1250) — matrix operand order repaired and checked by reading, not compiled, not run on hardware
//
#include <hip/hip_runtime.h>
#include <cstdint>

// ---------------------------------------------------------------------------
// BarlowTwins-style fused contrastive CE on gfx1250 (MI455X).
//   logits[b] = z1[b]^T z2[b]  (2048x2048, K=256), column logsumexp over n,
//   out = -mean_{b,m}( logits[b,m,m] - lse_n(logits[b,:,m]) )
// GEMM runs on v_wmma_f32_16x16x32_bf16 with 2-term bf16 error-compensated
// split (hi*hi + hi*lo + lo*hi) for ~fp32 accuracy at bf16 matrix throughput.
// ---------------------------------------------------------------------------

typedef __attribute__((ext_vector_type(16))) __bf16        v16bf;
typedef __attribute__((ext_vector_type(8)))  float         v8f;
typedef __attribute__((ext_vector_type(4)))  unsigned int  u32x4;
typedef __attribute__((ext_vector_type(2)))  unsigned int  u32x2;

constexpr int S      = 256;    // contraction (s) length
constexpr int N      = 2048;   // n == m == 2048
constexpr int BLK_M  = 64;     // m-columns per workgroup
constexpr int TILE_N = 128;    // n-rows per streamed tile
constexpr int KC     = 32;     // K chunk per LDS stage (== one WMMA frag K)
constexpr int KS     = 40;     // padded LDS row stride in bf16 (bank-safe)
constexpr int NT     = N / TILE_N;  // 16 n-tiles
constexpr int NKC    = S / KC;      // 8 k-chunks

union Frag  { v16bf v; u32x4 q[2]; };
union Pack4 { __bf16 h[4]; u32x2 u; };

__global__ __launch_bounds__(256) void bt_gemm_lse(
    const float* __restrict__ z1, const float* __restrict__ z2,
    float* __restrict__ partial)
{
    __shared__ __align__(16) __bf16 sAhi[TILE_N * KS];
    __shared__ __align__(16) __bf16 sAlo[TILE_N * KS];
    __shared__ __align__(16) __bf16 sBhi[BLK_M * KS];
    __shared__ __align__(16) __bf16 sBlo[BLK_M * KS];
    __shared__ float s_runmax[BLK_M], s_runsum[BLK_M];
    __shared__ float s_newmax[BLK_M], s_diag[BLK_M];
    __shared__ float s_pmax[4][BLK_M], s_psum[4][BLK_M];

    const int tid  = threadIdx.x;
    const int blk  = blockIdx.x;
    const int bIdx = blk >> 5;          // batch (8)
    const int mblk = blk & 31;          // m-block (32)
    const int m0   = mblk * BLK_M;
    const int diag_tile = mblk >> 1;    // n-tile containing the diagonal
    const int diag_off  = (mblk & 1) * 64;

    const float* baseA = z1 + bIdx * S * N;
    const float* baseB = z2 + bIdx * S * N;

    if (tid < BLK_M) {
        s_runmax[tid] = -__builtin_inff();
        s_runsum[tid] = 0.0f;
        s_diag[tid]   = 0.0f;
    }

    const int lane   = tid & 31;
    const int wid    = tid >> 5;
    const int wave_n = wid & 3;    // 4 row groups of 32
    const int wave_m = wid >> 2;   // 2 col groups of 32
    const int l15    = lane & 15;
    const int lh     = lane >> 4;

    // staging assignments
    const int a_n  = tid & 127, a_kh = tid >> 7;   // A: 128 n x 32 k
    const int b_m  = tid & 63,  b_kh = tid >> 6;   // B: 64 m  x 32 k

    for (int tile = 0; tile < NT; ++tile) {
        const int n0 = tile * TILE_N;
        v8f acc[2][2];
        const v8f zero8 = {0.f,0.f,0.f,0.f,0.f,0.f,0.f,0.f};
        acc[0][0] = zero8; acc[0][1] = zero8; acc[1][0] = zero8; acc[1][1] = zero8;

        for (int kc = 0; kc < NKC; ++kc) {
            const int s0 = kc * KC;
            __syncthreads();   // previous chunk's LDS reads complete

            // --- stage A chunk: z1[b][s0..s0+31][n0..n0+127] -> [n][k] hi/lo
            {
                const int kb = a_kh * 16;
                #pragma unroll
                for (int g = 0; g < 4; ++g) {
                    const int k = kb + g * 4;
                    const float* p = baseA + (s0 + k) * N + n0 + a_n;
                    Pack4 Ph, Pl;
                    #pragma unroll
                    for (int j = 0; j < 4; ++j) {
                        float x = p[j * N];
                        __bf16 hi = (__bf16)x;
                        Ph.h[j] = hi;
                        Pl.h[j] = (__bf16)(x - (float)hi);
                    }
                    *(u32x2*)(&sAhi[a_n * KS + k]) = Ph.u;
                    *(u32x2*)(&sAlo[a_n * KS + k]) = Pl.u;
                }
            }
            // --- stage B chunk: z2[b][s0..s0+31][m0..m0+63] -> [m][k] hi/lo
            {
                const int kb = b_kh * 8;
                #pragma unroll
                for (int g = 0; g < 2; ++g) {
                    const int k = kb + g * 4;
                    const float* p = baseB + (s0 + k) * N + m0 + b_m;
                    Pack4 Ph, Pl;
                    #pragma unroll
                    for (int j = 0; j < 4; ++j) {
                        float x = p[j * N];
                        __bf16 hi = (__bf16)x;
                        Ph.h[j] = hi;
                        Pl.h[j] = (__bf16)(x - (float)hi);
                    }
                    *(u32x2*)(&sBhi[b_m * KS + k]) = Ph.u;
                    *(u32x2*)(&sBlo[b_m * KS + k]) = Pl.u;
                }
            }
            __syncthreads();

            // --- fragments (LDS rows are fragment-native: 2x b128 each)
            Frag Ah[2], Al[2], Bh[2], Bl[2];
            #pragma unroll
            for (int sn = 0; sn < 2; ++sn) {
                const int row = wave_n * 32 + sn * 16 + l15;
                const __bf16* pa  = &sAhi[row * KS + lh * 8];
                const __bf16* pal = &sAlo[row * KS + lh * 8];
                Ah[sn].q[0] = *(const u32x4*)(pa);
                Ah[sn].q[1] = *(const u32x4*)(pa + 16);
                Al[sn].q[0] = *(const u32x4*)(pal);
                Al[sn].q[1] = *(const u32x4*)(pal + 16);
            }
            #pragma unroll
            for (int sm = 0; sm < 2; ++sm) {
                const int col = wave_m * 32 + sm * 16 + l15;
                const __bf16* pb  = &sBhi[col * KS + lh * 8];
                const __bf16* pbl = &sBlo[col * KS + lh * 8];
                Bh[sm].q[0] = *(const u32x4*)(pb);
                Bh[sm].q[1] = *(const u32x4*)(pb + 16);
                Bl[sm].q[0] = *(const u32x4*)(pbl);
                Bl[sm].q[1] = *(const u32x4*)(pbl + 16);
            }

            // --- error-compensated bf16 WMMA: hi*hi + hi*lo + lo*hi
            #pragma unroll
            for (int sn = 0; sn < 2; ++sn) {
                #pragma unroll
                for (int sm = 0; sm < 2; ++sm) {
                    v8f c = acc[sn][sm];
                    c = __builtin_amdgcn_wmma_f32_16x16x32_bf16(
                            false, Ah[sn].v, false, Bh[sm].v, (short)0, c, false, false);
                    c = __builtin_amdgcn_wmma_f32_16x16x32_bf16(
                            false, Ah[sn].v, false, Bl[sm].v, (short)0, c, false, false);
                    c = __builtin_amdgcn_wmma_f32_16x16x32_bf16(
                            false, Al[sn].v, false, Bh[sm].v, (short)0, c, false, false);
                    acc[sn][sm] = c;
                }
            }
        }

        // ---- online log-sum-exp update (columns = m), diagonal capture ----
        // C layout: VGPR v -> row v + 8*lh, col = l15.
        #pragma unroll
        for (int sm = 0; sm < 2; ++sm) {
            float mx = acc[0][sm][0];
            #pragma unroll
            for (int sn = 0; sn < 2; ++sn)
                #pragma unroll
                for (int vg = 0; vg < 8; ++vg)
                    mx = fmaxf(mx, acc[sn][sm][vg]);
            mx = fmaxf(mx, __shfl_xor(mx, 16, 32));
            if (lane < 16) s_pmax[wave_n][wave_m * 32 + sm * 16 + l15] = mx;
        }
        if (tile == diag_tile) {
            #pragma unroll
            for (int sn = 0; sn < 2; ++sn) {
                #pragma unroll
                for (int sm = 0; sm < 2; ++sm) {
                    const int c  = wave_m * 32 + sm * 16 + l15;
                    const int vg = diag_off + c - (wave_n * 32 + sn * 16 + 8 * lh);
                    if (vg >= 0 && vg < 8) s_diag[c] = acc[sn][sm][vg];
                }
            }
        }
        __syncthreads();
        if (tid < BLK_M) {
            float tm = fmaxf(fmaxf(s_pmax[0][tid], s_pmax[1][tid]),
                             fmaxf(s_pmax[2][tid], s_pmax[3][tid]));
            s_newmax[tid] = fmaxf(s_runmax[tid], tm);
        }
        __syncthreads();
        #pragma unroll
        for (int sm = 0; sm < 2; ++sm) {
            const int c = wave_m * 32 + sm * 16 + l15;
            const float nm = s_newmax[c];
            float sum = 0.0f;
            #pragma unroll
            for (int sn = 0; sn < 2; ++sn)
                #pragma unroll
                for (int vg = 0; vg < 8; ++vg)
                    sum += __expf(acc[sn][sm][vg] - nm);
            sum += __shfl_xor(sum, 16, 32);
            if (lane < 16) s_psum[wave_n][c] = sum;
        }
        __syncthreads();
        if (tid < BLK_M) {
            const float nm = s_newmax[tid];
            float rs = s_runsum[tid] * __expf(s_runmax[tid] - nm);
            rs += s_psum[0][tid] + s_psum[1][tid] + s_psum[2][tid] + s_psum[3][tid];
            s_runsum[tid] = rs;
            s_runmax[tid] = nm;
        }
    }

    __syncthreads();
    if (tid < BLK_M) {
        // logp[m,m] = diag - (max + log(sumexp))
        s_pmax[0][tid] = s_diag[tid] - s_runmax[tid] - __logf(s_runsum[tid]);
    }
    __syncthreads();
    if (tid == 0) {
        float sum = 0.0f;
        for (int i = 0; i < BLK_M; ++i) sum += s_pmax[0][i];
        partial[blk] = sum;
    }
}

__global__ __launch_bounds__(256) void bt_finalize(
    const float* __restrict__ partial, float* __restrict__ out)
{
    __shared__ float sh[256];
    const int t = threadIdx.x;
    sh[t] = partial[t];
    __syncthreads();
    for (int s = 128; s > 0; s >>= 1) {
        if (t < s) sh[t] += sh[t + s];
        __syncthreads();
    }
    if (t == 0) out[0] = -sh[0] * (1.0f / 16384.0f);   // -mean over 8*2048
}

extern "C" void kernel_launch(void* const* d_in, const int* in_sizes, int n_in,
                              void* d_out, int out_size, void* d_ws, size_t ws_size,
                              hipStream_t stream) {
    (void)in_sizes; (void)n_in; (void)out_size; (void)ws_size;
    const float* z1 = (const float*)d_in[0];
    const float* z2 = (const float*)d_in[1];
    float* out = (float*)d_out;
    float* ws  = (float*)d_ws;   // 256 floats of partials

    bt_gemm_lse<<<dim3(256), dim3(256), 0, stream>>>(z1, z2, ws);
    bt_finalize<<<dim3(1), dim3(256), 0, stream>>>(ws, out);
}
